// EdgePooling_7078106103935
// MI455X (gfx1250) — compile-verified, matrix operand order checked by reading, not run on hardware
//
#include <hip/hip_runtime.h>
#include <hip/hip_bf16.h>
#include <stdint.h>

typedef unsigned int u32;
typedef unsigned long long u64;
typedef float v2f  __attribute__((ext_vector_type(2)));
typedef float v8f  __attribute__((ext_vector_type(8)));
typedef _Float16 v16h __attribute__((ext_vector_type(16)));

#define RT 256           // radix threads per block
#define RK 16            // elements per thread in radix scatter
#define RCH (RT * RK)    // chunk per radix block
#define SCAN_ELEMS 1024  // elements per scan block
#define MAX_MATCH_ITERS 64

static __device__ __forceinline__ float ep_sigmoid(float z) {
  return 1.0f / (1.0f + expf(-z));
}

static __device__ __forceinline__ u64 ep_edge_rank(u32 rr, u32 rc, int n, int m, int e) {
  u64 mn = rr < rc ? rr : rc;
  u64 mx = rr < rc ? rc : rr;
  return (mn * (u64)n + mx) * (u64)m + (u64)e;
}

// ---------------------------------------------------------------------------
// 1. score = sigmoid(x @ w + b) via V_WMMA_F32_16X16X4_F32
//    One wave computes 16 rows. A = 16x4 f32 tile of x (ISA layout: lane<16
//    holds K={0,1}, lane>=16 holds K={2,3}); B = 4x16 with w broadcast over
//    all 16 columns, so every column of C holds the same 16 dot products.
// ---------------------------------------------------------------------------
__global__ void ep_score(const float* __restrict__ x, const float* __restrict__ w,
                         const float* __restrict__ bias, float* __restrict__ score,
                         int n, int d, int tiles) {
  int lane = threadIdx.x & 31;
  int wave = threadIdx.x >> 5;
  int t = blockIdx.x * (blockDim.x >> 5) + wave;
  if (t >= tiles) return;
  int half = (lane >> 4) & 1;
  int row = t * 16 + (lane & 15);
  int rclamp = row < n ? row : n - 1;
  const float* xrow = x + (size_t)rclamp * d;
  v8f acc = {0.f, 0.f, 0.f, 0.f, 0.f, 0.f, 0.f, 0.f};
#if __has_builtin(__builtin_amdgcn_wmma_f32_16x16x4_f32)
  for (int k = 0; k < d; k += 4) {
    __builtin_prefetch(xrow + k + 64, 0, 1);     // global_prefetch_b8
    v2f a = *(const v2f*)(xrow + k + half * 2);  // A[row][k+half*2 .. +1]
    v2f b = *(const v2f*)(w + k + half * 2);     // B[k..][*] = w broadcast
    acc = __builtin_amdgcn_wmma_f32_16x16x4_f32(false, a, false, b,
                                                (short)0, acc, false, false);
  }
#else
  // Fallback: f16 WMMA (codegen-confirmed builtin). 16-bit A 16x32 layout:
  // element e of lane maps to K = ((e>>3)<<4) + half*8 + (e&7).
  for (int k = 0; k < d; k += 32) {
    v16h a, b;
#pragma unroll
    for (int e = 0; e < 16; ++e) {
      int kk = ((e >> 3) << 4) + half * 8 + (e & 7);
      a[e] = (_Float16)xrow[k + kk];
      b[e] = (_Float16)w[k + kk];
    }
    acc = __builtin_amdgcn_wmma_f32_16x16x32_f16(false, a, false, b,
                                                 (short)0, acc, false, false);
  }
#endif
  if ((lane & 15) == 0) {  // lane 0 -> rows 0..7, lane 16 -> rows 8..15
    float bb = bias[0];
    int mbase = half ? 8 : 0;
#pragma unroll
    for (int j = 0; j < 8; ++j) {
      int r2 = t * 16 + mbase + j;
      if (r2 < n) score[r2] = ep_sigmoid(acc[j] + bb);
    }
  }
}

// ---------------------------------------------------------------------------
// Utility fills
// ---------------------------------------------------------------------------
__global__ void ep_fill_u32(u32* p, u32 v, long long n) {
  long long i = (long long)blockIdx.x * blockDim.x + threadIdx.x;
  if (i < n) p[i] = v;
}
__global__ void ep_fill_f32(float* p, float v, long long n) {
  long long i = (long long)blockIdx.x * blockDim.x + threadIdx.x;
  if (i < n) p[i] = v;
}
__global__ void ep_fill_u64(u64* p, u64 v, long long n) {
  long long i = (long long)blockIdx.x * blockDim.x + threadIdx.x;
  if (i < n) p[i] = v;
}

// ---------------------------------------------------------------------------
// 2. Stable LSD radix sort, 4-bit digits, u64 key + u32 payload.
// ---------------------------------------------------------------------------
__global__ void ep_radix_count(const u64* __restrict__ keys, int n, int shift,
                               u32* __restrict__ blockHist) {
  __shared__ u32 cnt[16];
  if (threadIdx.x < 16) cnt[threadIdx.x] = 0;
  __syncthreads();
  int base = blockIdx.x * RCH;
  for (int i = threadIdx.x; i < RCH; i += blockDim.x) {
    int idx = base + i;
    if (idx < n) atomicAdd(&cnt[(u32)(keys[idx] >> shift) & 15u], 1u);
  }
  __syncthreads();
  if (threadIdx.x < 16) blockHist[blockIdx.x * 16 + threadIdx.x] = cnt[threadIdx.x];
}

__global__ void ep_radix_scan(const u32* __restrict__ blockHist, u32* __restrict__ blockOff,
                              u32* __restrict__ digitBase, int nb) {
  __shared__ u32 tot[16];
  int d = threadIdx.x;
  if (d < 16) {
    u32 run = 0;
    for (int b = 0; b < nb; ++b) {
      u32 c = blockHist[b * 16 + d];
      blockOff[b * 16 + d] = run;
      run += c;
    }
    tot[d] = run;
  }
  __syncthreads();
  if (d == 0) {
    u32 run = 0;
    for (int i = 0; i < 16; ++i) { digitBase[i] = run; run += tot[i]; }
  }
}

__global__ void ep_radix_scatter(const u64* __restrict__ kin, const u32* __restrict__ vin,
                                 u64* __restrict__ kout, u32* __restrict__ vout,
                                 int n, int shift, const u32* __restrict__ blockOff,
                                 const u32* __restrict__ digitBase) {
  __shared__ u32 sh[RT * 16];  // 16 KB
  int t = threadIdx.x;
  int base = blockIdx.x * RCH + t * RK;  // thread owns contiguous RK elems -> stable
  u32 cnt[16];
#pragma unroll
  for (int i = 0; i < 16; ++i) cnt[i] = 0;
  u64 k[RK];
#pragma unroll
  for (int j = 0; j < RK; ++j) {
    int idx = base + j;
    if (idx < n) { k[j] = kin[idx]; cnt[(u32)(k[j] >> shift) & 15u]++; }
  }
#pragma unroll
  for (int i = 0; i < 16; ++i) sh[t * 16 + i] = cnt[i];
  __syncthreads();
  if (t < 16) {  // exclusive scan over threads for digit t
    u32 run = 0;
    for (int b = 0; b < RT; ++b) {
      u32 c = sh[b * 16 + t];
      sh[b * 16 + t] = run;
      run += c;
    }
  }
  __syncthreads();
  u32 off[16];
#pragma unroll
  for (int i = 0; i < 16; ++i)
    off[i] = digitBase[i] + blockOff[blockIdx.x * 16 + i] + sh[t * 16 + i];
#pragma unroll
  for (int j = 0; j < RK; ++j) {
    int idx = base + j;
    if (idx < n) {
      u32 dg = (u32)(k[j] >> shift) & 15u;
      u32 dst = off[dg]++;
      kout[dst] = k[j];
      vout[dst] = vin[idx];
    }
  }
}

// ---------------------------------------------------------------------------
// 3. Two-level inclusive scan (in place)
// ---------------------------------------------------------------------------
__global__ void ep_scan_local(u32* data, int n, u32* bsums) {
  __shared__ u32 sh[256];
  int t = threadIdx.x;
  int base = blockIdx.x * SCAN_ELEMS + t * 4;
  u32 v[4];
  u32 s = 0;
#pragma unroll
  for (int j = 0; j < 4; ++j) {
    int idx = base + j;
    v[j] = (idx < n) ? data[idx] : 0u;
    s += v[j];
  }
  sh[t] = s;
  __syncthreads();
  for (int off = 1; off < 256; off <<= 1) {
    u32 x = (t >= off) ? sh[t - off] : 0u;
    __syncthreads();
    sh[t] += x;
    __syncthreads();
  }
  u32 run = (t > 0) ? sh[t - 1] : 0u;
#pragma unroll
  for (int j = 0; j < 4; ++j) {
    int idx = base + j;
    run += v[j];
    if (idx < n) data[idx] = run;
  }
  if (t == 255) bsums[blockIdx.x] = sh[255];
}

__global__ void ep_scan_bsums(u32* bsums, int nb) {  // serial exclusive scan
  if (threadIdx.x == 0 && blockIdx.x == 0) {
    u32 carry = 0;
    for (int i = 0; i < nb; ++i) { u32 t = bsums[i]; bsums[i] = carry; carry += t; }
  }
}

__global__ void ep_scan_add(u32* data, int n, const u32* bsums) {
  u32 add = bsums[blockIdx.x];
  int base = blockIdx.x * SCAN_ELEMS + threadIdx.x * 4;
#pragma unroll
  for (int j = 0; j < 4; ++j) {
    int idx = base + j;
    if (idx < n) data[idx] += add;
  }
}

// ---------------------------------------------------------------------------
// 4. Ranking helpers
// ---------------------------------------------------------------------------
__global__ void ep_score_keys(const float* __restrict__ score, u64* __restrict__ keys,
                              u32* __restrict__ vals, int n) {
  int i = blockIdx.x * blockDim.x + threadIdx.x;
  if (i >= n) return;
  u32 u = __float_as_uint(score[i]);
  u ^= (u & 0x80000000u) ? 0xFFFFFFFFu : 0x80000000u;  // monotone flip
  keys[i] = (u64)u;
  vals[i] = (u32)i;
}

__global__ void ep_rank_from_order(const u32* __restrict__ order, u32* __restrict__ rank, int n) {
  int i = blockIdx.x * blockDim.x + threadIdx.x;
  if (i < n) rank[order[i]] = (u32)i;
}

// ---------------------------------------------------------------------------
// 5. Greedy maximal matching (flags[0]=done, flags[1]=candidate count)
// ---------------------------------------------------------------------------
__global__ void ep_match_reset(u64* node_min, int n, u32* flags) {
  int i = blockIdx.x * blockDim.x + threadIdx.x;
  if (i < n) node_min[i] = ~0ull;
  if (blockIdx.x == 0 && threadIdx.x == 0) flags[1] = 0;
}

__global__ void ep_match_min(const long long* __restrict__ ei, const u32* __restrict__ rank,
                             const u32* __restrict__ avail, u64* __restrict__ node_min,
                             u32* __restrict__ flags, int m, int n) {
  __shared__ u32 sc;
  if (threadIdx.x == 0) sc = 0;
  __syncthreads();
  bool done = flags[0] != 0;
  int e = blockIdx.x * blockDim.x + threadIdx.x;
  if (!done && e < m) {
    long long r = ei[e], c = ei[(size_t)m + e];
    if (r != c && avail[r] && avail[c]) {
      u64 er = ep_edge_rank(rank[r], rank[c], n, m, e);
      atomicMin(&node_min[r], er);
      atomicMin(&node_min[c], er);
      atomicAdd(&sc, 1u);
    }
  }
  __syncthreads();
  if (!done && threadIdx.x == 0 && sc) atomicAdd(&flags[1], sc);
}

__global__ void ep_match_check(u32* flags) {
  if (flags[0] == 0 && flags[1] == 0) flags[0] = 1;
}

__global__ void ep_match_select(const long long* __restrict__ ei, const u32* __restrict__ rank,
                                u32* __restrict__ avail, const u64* __restrict__ node_min,
                                u32* __restrict__ match, const u32* __restrict__ flags,
                                int m, int n) {
  if (flags[0]) return;
  int e = blockIdx.x * blockDim.x + threadIdx.x;
  if (e >= m) return;
  long long r = ei[e], c = ei[(size_t)m + e];
  if (r == c) return;
  if (!(avail[r] && avail[c])) return;
  u64 er = ep_edge_rank(rank[r], rank[c], n, m, e);
  if (er == node_min[r] && er == node_min[c]) {  // winner: unique per node
    match[e] = 1u;
    avail[r] = 0u;
    avail[c] = 0u;
  }
}

// ---------------------------------------------------------------------------
// 6. Clustering
// ---------------------------------------------------------------------------
__global__ void ep_iota(u32* p, int n) {
  int i = blockIdx.x * blockDim.x + threadIdx.x;
  if (i < n) p[i] = (u32)i;
}
__global__ void ep_apply_match(const long long* __restrict__ ei, const u32* __restrict__ match,
                               u32* __restrict__ cluster_raw, int m) {
  int e = blockIdx.x * blockDim.x + threadIdx.x;
  if (e < m && match[e]) cluster_raw[ei[(size_t)m + e]] = (u32)ei[e];
}
__global__ void ep_mark_present(const u32* __restrict__ cluster_raw, u32* __restrict__ newid, int n) {
  int i = blockIdx.x * blockDim.x + threadIdx.x;
  if (i < n) newid[cluster_raw[i]] = 1u;  // races write identical value
}
__global__ void ep_relabel(const u32* __restrict__ cluster_raw, const u32* __restrict__ newid,
                           u32* __restrict__ cluster, int n) {
  int i = blockIdx.x * blockDim.x + threadIdx.x;
  if (i < n) cluster[i] = newid[cluster_raw[i]] - 1u;
}

// ---------------------------------------------------------------------------
// 7. Pooling
// ---------------------------------------------------------------------------
__global__ void ep_pool_xs(const float* __restrict__ x, const float* __restrict__ score,
                           const u32* __restrict__ cluster, float* __restrict__ xs,
                           long long total, int d) {
  long long i = (long long)blockIdx.x * blockDim.x + threadIdx.x;
  if (i >= total) return;
  int node = (int)(i / d);
  int j = (int)(i - (long long)node * d);
  atomicAdd(&xs[(size_t)cluster[node] * d + j], x[i] * score[node]);
}
__global__ void ep_pool_norm(const float* __restrict__ score, const u32* __restrict__ cluster,
                             float* __restrict__ norm, int n) {
  int i = blockIdx.x * blockDim.x + threadIdx.x;
  if (i < n) atomicAdd(&norm[cluster[i]], score[i] * score[i]);
}
__global__ void ep_nscore(const float* __restrict__ score, const u32* __restrict__ cluster,
                          const float* __restrict__ norm, float* __restrict__ nscore, int n) {
  int i = blockIdx.x * blockDim.x + threadIdx.x;
  if (i < n) nscore[i] = score[i] / norm[cluster[i]];
}

// ---------------------------------------------------------------------------
// 8. Coalesce
// ---------------------------------------------------------------------------
__global__ void ep_val_keys(const long long* __restrict__ ei, const float* __restrict__ ea,
                            const float* __restrict__ nscore, const u32* __restrict__ cluster,
                            float* __restrict__ val, u64* __restrict__ keys,
                            u32* __restrict__ vals, int m, int n) {
  int e = blockIdx.x * blockDim.x + threadIdx.x;
  if (e >= m) return;
  long long r = ei[e], c = ei[(size_t)m + e];
  val[e] = nscore[r] * ea[e] * nscore[c];
  keys[e] = (u64)cluster[r] * (u64)(n + 1) + (u64)cluster[c];
  vals[e] = (u32)e;
}
__global__ void ep_mark_first(const u64* __restrict__ sk, u32* __restrict__ g, int m) {
  int j = blockIdx.x * blockDim.x + threadIdx.x;
  if (j < m) g[j] = (j == 0 || sk[j] != sk[j - 1]) ? 1u : 0u;
}
__global__ void ep_coalesce_sum(const u32* __restrict__ perm, const u32* __restrict__ g,
                                const float* __restrict__ val, float* __restrict__ sums, int m) {
  int j = blockIdx.x * blockDim.x + threadIdx.x;
  if (j < m) atomicAdd(&sums[g[j] - 1u], val[perm[j]]);
}
__global__ void ep_final_edges(const u64* __restrict__ sk, const u32* __restrict__ g,
                               const float* __restrict__ sums, float* __restrict__ out,
                               size_t o_row, size_t o_col, size_t o_val, int m, int n) {
  int j = blockIdx.x * blockDim.x + threadIdx.x;
  if (j >= m) return;
  u64 k = sk[j];
  bool first = (j == 0) || (k != sk[j - 1]);
  u64 np1 = (u64)(n + 1);
  out[o_row + j] = first ? (float)(k / np1) : (float)n;
  out[o_col + j] = first ? (float)(k % np1) : (float)n;
  out[o_val + j] = first ? sums[g[j] - 1u] : 0.0f;
}

// ---------------------------------------------------------------------------
// 9. Output writers (everything flattened to f32)
// ---------------------------------------------------------------------------
__global__ void ep_write_u32_as_f32(const u32* __restrict__ src, float* __restrict__ out,
                                    size_t off, int n) {
  int i = blockIdx.x * blockDim.x + threadIdx.x;
  if (i < n) out[off + i] = (float)src[i];
}
__global__ void ep_write_f32(const float* __restrict__ src, float* __restrict__ out,
                             size_t off, int n) {
  int i = blockIdx.x * blockDim.x + threadIdx.x;
  if (i < n) out[off + i] = src[i];
}
__global__ void ep_write_c(const u32* __restrict__ newid, float* __restrict__ out,
                           size_t off, int n) {
  out[off] = (float)newid[n - 1];
}

// ---------------------------------------------------------------------------
// Host helpers (launch-only; graph-capture safe)
// ---------------------------------------------------------------------------
static inline unsigned int gceil(long long x, int b) { return (unsigned int)((x + b - 1) / b); }

static void ep_radix_sort(u64*& ka, u32*& va, u64*& kb, u32*& vb, int n, int passes,
                          u32* bh, u32* bo, u32* db, hipStream_t s) {
  int nb = (n + RCH - 1) / RCH;
  for (int p = 0; p < passes; ++p) {
    int shift = p * 4;
    ep_radix_count<<<nb, RT, 0, s>>>(ka, n, shift, bh);
    ep_radix_scan<<<1, 64, 0, s>>>(bh, bo, db, nb);
    ep_radix_scatter<<<nb, RT, 0, s>>>(ka, va, kb, vb, n, shift, bo, db);
    u64* tk = ka; ka = kb; kb = tk;
    u32* tv = va; va = vb; vb = tv;
  }
}

static void ep_scan(u32* data, int n, u32* bsums, hipStream_t s) {
  int nb = (n + SCAN_ELEMS - 1) / SCAN_ELEMS;
  ep_scan_local<<<nb, 256, 0, s>>>(data, n, bsums);
  ep_scan_bsums<<<1, 1, 0, s>>>(bsums, nb);
  ep_scan_add<<<nb, 256, 0, s>>>(data, n, bsums);
}

extern "C" void kernel_launch(void* const* d_in, const int* in_sizes, int n_in,
                              void* d_out, int out_size, void* d_ws, size_t ws_size,
                              hipStream_t stream) {
  const float*     x    = (const float*)d_in[0];
  const long long* ei   = (const long long*)d_in[1];
  const float*     ea   = (const float*)d_in[2];
  const float*     w    = (const float*)d_in[3];
  const float*     bias = (const float*)d_in[4];
  int d = in_sizes[3];
  int n = in_sizes[0] / d;
  int m = in_sizes[1] / 2;
  float* out = (float*)d_out;

  // Output layout (reference return order, flattened to f32)
  size_t o_xs = 0;
  size_t o_row = (size_t)n * d;
  size_t o_col = o_row + (size_t)m;
  size_t o_val = o_row + (size_t)2 * m;
  size_t o_cl  = o_val + (size_t)m;
  size_t o_mt  = o_cl + (size_t)n;
  size_t o_sc  = o_mt + (size_t)m;
  size_t o_c   = o_sc + (size_t)n;

  // Carve workspace
  char* p = (char*)d_ws;
  auto carve = [&](size_t bytes) -> char* {
    char* r = p;
    p += (bytes + 255) & ~(size_t)255;
    return r;
  };
  int nbMax = (m + RCH - 1) / RCH;
  float* score       = (float*)carve(sizeof(float) * n);
  u64*   kbuf0       = (u64*)carve(sizeof(u64) * m);
  u64*   kbuf1       = (u64*)carve(sizeof(u64) * m);
  u32*   vbuf0       = (u32*)carve(sizeof(u32) * m);
  u32*   vbuf1       = (u32*)carve(sizeof(u32) * m);
  u32*   rank        = (u32*)carve(sizeof(u32) * n);
  u64*   node_min    = (u64*)carve(sizeof(u64) * n);
  u32*   avail       = (u32*)carve(sizeof(u32) * n);
  u32*   match       = (u32*)carve(sizeof(u32) * m);
  u32*   flags       = (u32*)carve(sizeof(u32) * 16);
  u32*   cluster_raw = (u32*)carve(sizeof(u32) * n);
  u32*   cluster     = (u32*)carve(sizeof(u32) * n);
  u32*   newid       = (u32*)carve(sizeof(u32) * n);
  float* normv       = (float*)carve(sizeof(float) * n);
  float* nscore      = (float*)carve(sizeof(float) * n);
  float* val         = (float*)carve(sizeof(float) * m);
  u32*   gid         = (u32*)carve(sizeof(u32) * m);
  float* sums        = (float*)carve(sizeof(float) * m);
  u32*   blockHist   = (u32*)carve(sizeof(u32) * 16 * nbMax);
  u32*   blockOff    = (u32*)carve(sizeof(u32) * 16 * nbMax);
  u32*   digitBase   = (u32*)carve(sizeof(u32) * 16);
  u32*   bsums       = (u32*)carve(sizeof(u32) * 8192);

  // 1. WMMA scorer
  int tiles = (n + 15) / 16;
  ep_score<<<gceil((long long)tiles * 32, 256), 256, 0, stream>>>(x, w, bias, score, n, d, tiles);

  // 2. Rank = stable ascending sort of score (index tie-break from stability)
  ep_score_keys<<<gceil(n, 256), 256, 0, stream>>>(score, kbuf0, vbuf0, n);
  {
    u64* ka = kbuf0; u32* va = vbuf0; u64* kb = kbuf1; u32* vb = vbuf1;
    ep_radix_sort(ka, va, kb, vb, n, 8, blockHist, blockOff, digitBase, stream);  // 32 bits
    ep_rank_from_order<<<gceil(n, 256), 256, 0, stream>>>(va, rank, n);
  }

  // 3. Maximal matching (fixed trip count, device-side convergence flag)
  ep_fill_u32<<<1, 16, 0, stream>>>(flags, 0u, 16);
  ep_fill_u32<<<gceil(n, 256), 256, 0, stream>>>(avail, 1u, n);
  ep_fill_u32<<<gceil(m, 256), 256, 0, stream>>>(match, 0u, m);
  for (int it = 0; it < MAX_MATCH_ITERS; ++it) {
    ep_match_reset<<<gceil(n, 256), 256, 0, stream>>>(node_min, n, flags);
    ep_match_min<<<gceil(m, 256), 256, 0, stream>>>(ei, rank, avail, node_min, flags, m, n);
    ep_match_check<<<1, 1, 0, stream>>>(flags);
    ep_match_select<<<gceil(m, 256), 256, 0, stream>>>(ei, rank, avail, node_min, match, flags, m, n);
  }

  // 4. Clusters + consecutive relabel
  ep_iota<<<gceil(n, 256), 256, 0, stream>>>(cluster_raw, n);
  ep_apply_match<<<gceil(m, 256), 256, 0, stream>>>(ei, match, cluster_raw, m);
  ep_fill_u32<<<gceil(n, 256), 256, 0, stream>>>(newid, 0u, n);
  ep_mark_present<<<gceil(n, 256), 256, 0, stream>>>(cluster_raw, newid, n);
  ep_scan(newid, n, bsums, stream);  // inclusive cumsum(present)
  ep_relabel<<<gceil(n, 256), 256, 0, stream>>>(cluster_raw, newid, cluster, n);

  // 5. Pooled features xs and norm
  long long total = (long long)n * d;
  ep_fill_f32<<<gceil(total, 256), 256, 0, stream>>>(out + o_xs, 0.0f, total);
  ep_fill_f32<<<gceil(n, 256), 256, 0, stream>>>(normv, 0.0f, n);
  ep_pool_xs<<<gceil(total, 256), 256, 0, stream>>>(x, score, cluster, out + o_xs, total, d);
  ep_pool_norm<<<gceil(n, 256), 256, 0, stream>>>(score, cluster, normv, n);
  ep_nscore<<<gceil(n, 256), 256, 0, stream>>>(score, cluster, normv, nscore, n);

  // 6. Edge values + coalesce sort (keys < (n+1)^2 < 2^34 -> 9 x 4-bit passes)
  ep_val_keys<<<gceil(m, 256), 256, 0, stream>>>(ei, ea, nscore, cluster, val, kbuf0, vbuf0, m, n);
  u64* sk; u32* perm;
  {
    u64* ka = kbuf0; u32* va = vbuf0; u64* kb = kbuf1; u32* vb = vbuf1;
    ep_radix_sort(ka, va, kb, vb, m, 9, blockHist, blockOff, digitBase, stream);
    sk = ka; perm = va;
  }
  ep_mark_first<<<gceil(m, 256), 256, 0, stream>>>(sk, gid, m);
  ep_scan(gid, m, bsums, stream);  // gid[j] = group id + 1
  ep_fill_f32<<<gceil(m, 256), 256, 0, stream>>>(sums, 0.0f, m);
  ep_coalesce_sum<<<gceil(m, 256), 256, 0, stream>>>(perm, gid, val, sums, m);
  ep_final_edges<<<gceil(m, 256), 256, 0, stream>>>(sk, gid, sums, out, o_row, o_col, o_val, m, n);

  // 7. Remaining outputs
  ep_write_u32_as_f32<<<gceil(n, 256), 256, 0, stream>>>(cluster, out, o_cl, n);
  ep_write_u32_as_f32<<<gceil(m, 256), 256, 0, stream>>>(match, out, o_mt, m);
  ep_write_f32<<<gceil(n, 256), 256, 0, stream>>>(score, out, o_sc, n);
  ep_write_c<<<1, 1, 0, stream>>>(newid, out, o_c, n);
}
